// Selection_window_50087908606713
// MI455X (gfx1250) — compile-verified, matrix operand order checked by reading, not run on hardware
//
#include <hip/hip_runtime.h>
#include <hip/hip_bf16.h>

typedef __attribute__((ext_vector_type(16))) _Float16 v16h;
typedef __attribute__((ext_vector_type(8)))  _Float16 v8h;
typedef __attribute__((ext_vector_type(8)))  float    v8f;
typedef __attribute__((ext_vector_type(4)))  float    v4f;

#define WMMA_F16(a,b,c) __builtin_amdgcn_wmma_f32_16x16x32_f16(false,(a),false,(b),(short)0,(c),false,false)

union HU { v16h v; v8h h[2]; };

static __device__ inline v16h load_frag16(const _Float16* p) {
    HU u;
    u.h[0] = *(const v8h*)p;
    u.h[1] = *(const v8h*)(p + 8);
    return u.v;
}

// Copy 2KB (this wave's slice of a 16KB chunk) global->LDS via the CDNA5 async
// path (ASYNCcnt). Per lane: 4 x 16B. IOFFSET advances BOTH the global and the
// LDS address (ISA 15.18.3), so one vaddr/vdst pair covers all four.
static __device__ inline void async_copy_2k(unsigned ldsdst, unsigned voff, unsigned long long gbase) {
    asm volatile(
        "global_load_async_to_lds_b128 %0, %1, %2 offset:0\n\t"
        "global_load_async_to_lds_b128 %0, %1, %2 offset:512\n\t"
        "global_load_async_to_lds_b128 %0, %1, %2 offset:1024\n\t"
        "global_load_async_to_lds_b128 %0, %1, %2 offset:1536"
        :: "v"(ldsdst), "v"(voff), "s"(gbase)
        : "memory");
}

static __device__ inline void wait_asynccnt_le4() {
    asm volatile("s_wait_asynccnt 0x4" ::: "memory");
}

// ---- Weight prep: pack W1/W2 (fp32 row-major) into f16 WMMA-B fragments,
// grouped into 32 contiguous 16KB chunks: chunk 2u = W1 frags for hidden slab u
// (frag index t*2+half, t=0..7), chunk 2u+1 = W2 frags for k-step u (nt=0..15).
// Within a chunk: halves index = (frag*32 + lane)*16 + j,
//   lane column n = nt*16 + (lane&15),
//   K = t*32 + ((lane>=16)?8:0) + j + (j&8)   (16-bit B-matrix VGPR layout).
__global__ void prep_w1_kernel(const float* __restrict__ W, _Float16* __restrict__ wcomb) {
    int id = blockIdx.x * 256 + threadIdx.x;      // 131072 total
    int j    = id & 15;
    int lane = (id >> 4) & 31;
    int rest = id >> 9;                           // t*32 + nt
    int nt = rest & 31;
    int t  = rest >> 5;
    int base = (lane >> 4) << 3;
    int K = t * 32 + base + j + (j & 8);
    int n = nt * 16 + (lane & 15);
    int u = nt >> 1, half = nt & 1;
    size_t dst = (size_t)(2 * u) * 8192 + (size_t)((((t * 2 + half) * 32 + lane) << 4) + j);
    wcomb[dst] = (_Float16)W[(size_t)K * 512 + (size_t)n];
}

__global__ void prep_w2_kernel(const float* __restrict__ W, _Float16* __restrict__ wcomb) {
    int id = blockIdx.x * 256 + threadIdx.x;      // 131072 total
    int j    = id & 15;
    int lane = (id >> 4) & 31;
    int rest = id >> 9;                           // t*16 + nt
    int nt = rest & 15;
    int t  = rest >> 4;                           // = u
    int base = (lane >> 4) << 3;
    int K = t * 32 + base + j + (j & 8);
    int n = nt * 16 + (lane & 15);
    size_t dst = (size_t)(2 * t + 1) * 8192 + (size_t)(((nt * 32 + lane) << 4) + j);
    wcomb[dst] = (_Float16)W[(size_t)K * 256 + (size_t)n];
}

// limit[b] = (float) argmax(event_time[b, 0:4096])  (first occurrence)
__global__ void limit_kernel(const float* __restrict__ et, float* __restrict__ lim) {
    int b = blockIdx.x;
    int lane = threadIdx.x;            // 32 threads = 1 wave
    float best = -3.4e38f;
    int bi = 0;
    for (int l = lane; l < 4096; l += 32) {
        float v = et[b * 4096 + l];
        if (v > best) { best = v; bi = l; }
    }
    #pragma unroll
    for (int msk = 16; msk >= 1; msk >>= 1) {
        int pidx = ((lane ^ msk) << 2);
        float ov = __builtin_bit_cast(float,
            __builtin_amdgcn_ds_bpermute(pidx, __builtin_bit_cast(int, best)));
        int oi = __builtin_amdgcn_ds_bpermute(pidx, bi);
        if (ov > best || (ov == best && oi < bi)) { best = ov; bi = oi; }
    }
    if (lane == 0) lim[b] = (float)bi;
}

// Fused MLP (f16 WMMA, f32 accumulate) + offset + mask + interpolation gather.
// Block = 8 waves; each wave owns 16 rows. Weight fragments are staged in LDS
// by async copies (3 rotating 16KB buffers, one chunk in flight), shared by all
// 8 waves -> 8x less L2 fragment traffic than per-wave global loads.
__global__ __launch_bounds__(256, 1) void mlp_gather_kernel(
    const float* __restrict__ xin,      // (64, 4096, 256)
    const float* __restrict__ b1,       // (512)
    const float* __restrict__ b2,       // (256)
    const float* __restrict__ w3,       // (256)
    const _Float16* __restrict__ wcomb, // 32 x 16KB fragment chunks
    const float* __restrict__ lim,      // (8)
    float* __restrict__ outp,           // (65536, 256)
    float* __restrict__ maskp)          // (65536)
{
    __shared__ __align__(16) _Float16 wlds[3][8192];   // 3 x 16KB chunk buffers
    __shared__ __align__(16) _Float16 stage[8][16 * 40]; // per-wave 16x32 transpose buf
    __shared__ float offlds[8][16];

    const int tid   = threadIdx.x;
    const int wv    = tid >> 5;
    const int lane  = tid & 31;
    const int nlane = lane & 15;
    const int hi    = lane >> 4;
    const int kbase = hi * 8;

    const int rowbase = (blockIdx.x * 8 + wv) * 16;   // 16 rows per wave, one bk
    const int bk    = rowbase >> 10;
    const int b     = bk >> 3;
    const int sbase = rowbase & 1023;

    // Low 32 bits of a flat LDS pointer == LDS byte offset (ISA aperture rule).
    const unsigned ldsw  = (unsigned)(uintptr_t)&wlds[0][0];
    const unsigned slice = (unsigned)(wv * 2048 + lane * 16); // this lane's slice of a chunk
    const unsigned long long wbase = (unsigned long long)(uintptr_t)wcomb;

    // Kick off chunks 0 (W1 slab 0) and 1 (W2 kstep 0).
    async_copy_2k(ldsw + 0 * 16384 + slice, slice, wbase + 0ull * 16384ull);
    async_copy_2k(ldsw + 1 * 16384 + slice, slice, wbase + 1ull * 16384ull);

    // ---- A0 fragments: 16x256 activations, f32 -> f16, WMMA A layout.
    const float* rowp = xin + ((size_t)bk * 4096 + (size_t)(2 * (sbase + nlane))) * 256;
    v16h a0[8];
    #pragma unroll
    for (int t = 0; t < 8; ++t) {
        const float* rp = rowp + t * 32 + kbase;
        v4f q0 = *(const v4f*)(rp);
        v4f q1 = *(const v4f*)(rp + 4);
        v4f q2 = *(const v4f*)(rp + 16);
        v4f q3 = *(const v4f*)(rp + 20);
        v16h a;
        #pragma unroll
        for (int i = 0; i < 4; ++i) {
            a[i]      = (_Float16)q0[i];
            a[4 + i]  = (_Float16)q1[i];
            a[8 + i]  = (_Float16)q2[i];
            a[12 + i] = (_Float16)q3[i];
        }
        a0[t] = a;
    }

    // ---- GEMM2 accumulators, bias-initialized (C layout: lane = col n, vgpr r = row m)
    v8f acc2[16];
    #pragma unroll
    for (int nt = 0; nt < 16; ++nt) {
        float bv = b2[nt * 16 + nlane];
        v8f c;
        #pragma unroll
        for (int r = 0; r < 8; ++r) c[r] = bv;
        acc2[nt] = c;
    }

    _Float16* st = stage[wv];

    for (int u = 0; u < 16; ++u) {
        const int idxA = (2 * u) % 3;
        const int idxB = (2 * u + 1) % 3;

        // ===== phase A: GEMM1 on hidden slab u (chunk 2u) =====
        wait_asynccnt_le4();   // in-order completion => chunk 2u has landed
        __syncthreads();
        if (u < 15)            // prefetch W1 slab u+1 into the free buffer
            async_copy_2k(ldsw + (unsigned)(((2 * u + 2) % 3) * 16384) + slice,
                          slice, wbase + (unsigned long long)(2 * u + 2) * 16384ull);
        const _Float16* bA = &wlds[idxA][0];
        #pragma unroll
        for (int half = 0; half < 2; ++half) {
            float bv = b1[(2 * u + half) * 16 + nlane];
            v8f c1;
            #pragma unroll
            for (int r = 0; r < 8; ++r) c1[r] = bv;
            #pragma unroll
            for (int t = 0; t < 8; ++t) {
                v16h bf = load_frag16(bA + (((t * 2 + half) * 32 + lane) << 4));
                c1 = WMMA_F16(a0[t], bf, c1);
            }
            // ReLU + stage as f16 (transpose C layout -> A layout via LDS)
            #pragma unroll
            for (int r = 0; r < 8; ++r) {
                int mm = r + hi * 8;
                st[mm * 40 + half * 16 + nlane] = (_Float16)fmaxf(c1[r], 0.0f);
            }
        }
        // A1 fragment for GEMM2 (16x32 f16): two 16B ds loads per lane
        HU ua;
        ua.h[0] = *(const v8h*)(st + nlane * 40 + kbase);
        ua.h[1] = *(const v8h*)(st + nlane * 40 + kbase + 16);
        v16h a1 = ua.v;

        // ===== phase B: GEMM2 k-step u (chunk 2u+1) =====
        wait_asynccnt_le4();   // chunk 2u+1 has landed only for u<15 (at u=15 no later chunk is queued)
        __syncthreads();
        if (u < 15)            // prefetch W2 kstep u+1 (reuses slab-u's buffer)
            async_copy_2k(ldsw + (unsigned)(((2 * u + 3) % 3) * 16384) + slice,
                          slice, wbase + (unsigned long long)(2 * u + 3) * 16384ull);
        const _Float16* bB = &wlds[idxB][0];
        #pragma unroll
        for (int nt = 0; nt < 16; ++nt) {
            v16h bf = load_frag16(bB + ((nt * 32 + lane) << 4));
            acc2[nt] = WMMA_F16(a1, bf, acc2[nt]);
        }
    }

    // ---- ReLU + W3 dot -> per-row offset
    float off[8];
    #pragma unroll
    for (int r = 0; r < 8; ++r) off[r] = 0.0f;
    #pragma unroll
    for (int nt = 0; nt < 16; ++nt) {
        float w3v = w3[nt * 16 + nlane];
        #pragma unroll
        for (int r = 0; r < 8; ++r) off[r] += fmaxf(acc2[nt][r], 0.0f) * w3v;
    }
    // reduce across the 16 lanes of each half-wave
    #pragma unroll
    for (int msk = 8; msk >= 1; msk >>= 1) {
        int pidx = ((lane ^ msk) << 2);
        #pragma unroll
        for (int r = 0; r < 8; ++r) {
            off[r] += __builtin_bit_cast(float,
                __builtin_amdgcn_ds_bpermute(pidx, __builtin_bit_cast(int, off[r])));
        }
    }
    if (nlane == 0) {
        #pragma unroll
        for (int r = 0; r < 8; ++r) offlds[wv][r + hi * 8] = off[r];
    }

    // ---- epilogue: pos, mask, linear-interp gather (fp32, b128 vectorized)
    const float limv = lim[b];
    const float* xbk = xin + (size_t)bk * 4096 * 256;
    for (int mm = 0; mm < 16; ++mm) {
        float offm = offlds[wv][mm];
        int s = sbase + mm;
        float p = (float)(2 * s) + offm;
        if (lane == mm) maskp[rowbase + mm] = (p <= limv) ? 1.0f : 0.0f;
        float x0 = floorf(p);
        float w  = p - x0;
        int i0 = (int)x0;
        int i1 = i0 + 1;
        float f0 = (i0 >= 0 && i0 <= 4095) ? (1.0f - w) : 0.0f;
        float f1 = (i1 >= 0 && i1 <= 4095) ? w : 0.0f;
        int ic0 = i0 < 0 ? 0 : (i0 > 4095 ? 4095 : i0);
        int ic1 = i1 < 0 ? 0 : (i1 > 4095 ? 4095 : i1);
        const v4f* p0 = (const v4f*)(xbk + (size_t)ic0 * 256 + lane * 8);
        const v4f* p1 = (const v4f*)(xbk + (size_t)ic1 * 256 + lane * 8);
        v4f r0 = p0[0] * f0 + p1[0] * f1;
        v4f r1 = p0[1] * f0 + p1[1] * f1;
        float* op = outp + (size_t)(rowbase + mm) * 256 + (size_t)lane * 8;
        *(v4f*)op       = r0;
        *(v4f*)(op + 4) = r1;
    }
}

extern "C" void kernel_launch(void* const* d_in, const int* in_sizes, int n_in,
                              void* d_out, int out_size, void* d_ws, size_t ws_size,
                              hipStream_t stream) {
    const float* xin = (const float*)d_in[0];   // input   (8,8,4096,256)
    const float* et  = (const float*)d_in[1];   // event_time (8,4096)
    const float* W1  = (const float*)d_in[2];   // (256,512)
    const float* b1  = (const float*)d_in[3];   // (512)
    const float* W2  = (const float*)d_in[4];   // (512,256)
    const float* b2  = (const float*)d_in[5];   // (256)
    const float* W3  = (const float*)d_in[6];   // (256,1)
    (void)in_sizes; (void)n_in; (void)out_size; (void)ws_size;

    float* outp  = (float*)d_out;                       // 8*8*1024*256 fp32
    float* maskp = outp + (size_t)8 * 8 * 1024 * 256;   // 8*8*1024 fp32

    char* ws = (char*)d_ws;
    float*    lim   = (float*)ws;                       // 8 floats
    _Float16* wcomb = (_Float16*)(ws + 256);            // 32 x 16KB = 512KB

    prep_w1_kernel<<<512, 256, 0, stream>>>(W1, wcomb);
    prep_w2_kernel<<<512, 256, 0, stream>>>(W2, wcomb);
    limit_kernel<<<8, 32, 0, stream>>>(et, lim);
    mlp_gather_kernel<<<512, 256, 0, stream>>>(xin, b1, b2, W3, wcomb, lim, outp, maskp);
}
